// DeepseekV2DecoderLayer_49709951483963
// MI455X (gfx1250) — compile-verified, matrix operand order checked by reading, not run on hardware
//
#include <hip/hip_runtime.h>
#include <hip/hip_bf16.h>
#include <math.h>

// ---------------- problem constants ----------------
#define TTOK  4096
#define HDIM  2048
#define NH    16
#define DN    128
#define DR    64
#define DQK   192      // DN + DR
#define DV    128
#define QLR   1536
#define KVLR  512
#define INTER 10944
#define EPSF  1e-6f
#define SCALEF 0.07216878364870323f   // 192^-0.5

typedef _Float16 f16;
typedef __attribute__((ext_vector_type(16))) _Float16 v16h;
typedef __attribute__((ext_vector_type(8)))  _Float16 v8h;
typedef __attribute__((ext_vector_type(8)))  float    v8f;
typedef __attribute__((ext_vector_type(4)))  int      i32x4;

// -------- CDNA5 async global->LDS copy (guarded), with sync fallback --------
#if defined(__has_builtin)
#  if __has_builtin(__builtin_amdgcn_global_load_async_to_lds_b128)
#    define HAVE_ASYNC_LDS 1
#  endif
#  if __has_builtin(__builtin_amdgcn_s_wait_asynccnt)
#    define HAVE_WAIT_ASYNC 1
#  endif
#endif

__device__ __forceinline__ void cp16_g2l(f16* l, const f16* g) {
#if defined(HAVE_ASYNC_LDS)
  __builtin_amdgcn_global_load_async_to_lds_b128(
      (__attribute__((address_space(1))) i32x4*)g,
      (__attribute__((address_space(3))) i32x4*)l, 0, 0);
#else
  *(v8h*)l = *(const v8h*)g;
#endif
}

template <int N>
__device__ __forceinline__ void cp_commit() {
#if defined(HAVE_ASYNC_LDS)
#  if defined(HAVE_WAIT_ASYNC)
  __builtin_amdgcn_s_wait_asynccnt((short)N);
#  else
  asm volatile("s_wait_asynccnt %0" ::"i"(N) : "memory");
#  endif
#else
  asm volatile("s_wait_dscnt 0x0" ::: "memory");
#endif
}

// ---------------- WMMA helpers ----------------
// CDNA5 16x16x32 f16 fragment layout (ISA 7.12.2):
// A: lane L holds row (L%16); with h = L/16, halfs 0..7 are K = h*8..h*8+7,
//    halfs 8..15 are K = 16+h*8 .. 16+h*8+7.  Two contiguous 16B loads.
// B for C = A * W^T loads W rows with the identical pattern (lane -> col n).
// C/D: lane L holds col (L%16); VGPR v holds row v + 8*(L/16).
__device__ __forceinline__ v16h frag_ld(const f16* p, int h) {
  v8h lo = *(const v8h*)(p + h * 8);
  v8h hi = *(const v8h*)(p + 16 + h * 8);
  return __builtin_shufflevector(lo, hi, 0,1,2,3,4,5,6,7,8,9,10,11,12,13,14,15);
}
__device__ __forceinline__ v8f wmma16(v16h a, v16h b, v8f c) {
  return __builtin_amdgcn_wmma_f32_16x16x32_f16(false, a, false, b, (short)0, c,
                                                false, false);
}

// ---------------- utility kernels ----------------
__global__ void __launch_bounds__(256)
cvt_f32_f16(const float* __restrict__ s, f16* __restrict__ d, long long n) {
  long long i = (long long)blockIdx.x * blockDim.x + threadIdx.x;
  long long st = (long long)gridDim.x * blockDim.x;
  for (; i < n; i += st) d[i] = (f16)s[i];
}

template <typename TIn>
__global__ void __launch_bounds__(256)
rmsnorm_k(const TIn* __restrict__ x, const float* __restrict__ w,
          f16* __restrict__ out, int cols, int in_stride) {
  __shared__ float red[9];
  const int row = blockIdx.x;
  const TIn* xr = x + (size_t)row * in_stride;
  float ss = 0.f;
  for (int c = threadIdx.x; c < cols; c += 256) {
    float v = (float)xr[c];
    ss += v * v;
  }
  for (int o = 16; o; o >>= 1) ss += __shfl_down(ss, o, 32);
  if ((threadIdx.x & 31) == 0) red[threadIdx.x >> 5] = ss;
  __syncthreads();
  if (threadIdx.x == 0) {
    float t = 0.f;
    for (int i = 0; i < 8; ++i) t += red[i];
    red[8] = rsqrtf(t / (float)cols + EPSF);
  }
  __syncthreads();
  const float inv = red[8];
  for (int c = threadIdx.x; c < cols; c += 256)
    out[(size_t)row * cols + c] = (f16)((float)xr[c] * inv * w[c]);
}

// ------------- generic WMMA GEMM:  C[M,N] = A[M,K] * W[N,K]^T ----------------
// Block: 256 thr = 8 waves; wave computes 32Mx64N; block 256Mx64N.
// W tile (64 rows x 32 K halfs) staged in LDS, double-buffered via async copy.
// EPI: 0 -> f16 store, 1 -> f32 store, 2 -> f32 store of resid + acc
template <int EPI>
__global__ void __launch_bounds__(256)
gemm_wmma_k(const f16* __restrict__ A, const f16* __restrict__ W,
            void* __restrict__ outp, const float* __restrict__ resid,
            int N, int K) {
  __shared__ __attribute__((aligned(16))) f16 wt[2][64 * 32];
  const int tid = threadIdx.x;
  const int lane = tid & 31, wave = tid >> 5;
  const int m0 = blockIdx.y * 256 + wave * 32;
  const int n0 = blockIdx.x * 64;
  const int col = lane & 15, h = lane >> 4;
  const f16* a_row0 = A + (size_t)(m0 + col) * K;
  const f16* a_row1 = A + (size_t)(m0 + 16 + col) * K;
  // this thread's cooperative-staging slice: 256 thr x 16B = 4KB tile
  const int srow = tid >> 2, sseg = (tid & 3) * 8;
  const f16* g_src = W + (size_t)(n0 + srow) * K + sseg;
  f16* l_dst[2] = {&wt[0][srow * 32 + sseg], &wt[1][srow * 32 + sseg]};

  v8f acc0[4] = {}, acc1[4] = {};
  cp16_g2l(l_dst[0], g_src);  // stage k0 = 0
  int buf = 0;
  for (int k0 = 0; k0 < K; k0 += 32) {
    if (k0 + 32 < K) {
      cp16_g2l(l_dst[buf ^ 1], g_src + k0 + 32);  // stage next tile
      cp_commit<1>();                             // current tile has landed
    } else {
      cp_commit<0>();
    }
    __syncthreads();
    __builtin_prefetch(a_row0 + k0 + 256, 0, 0);
    v16h a0 = frag_ld(a_row0 + k0, h);
    v16h a1 = frag_ld(a_row1 + k0, h);
    const f16* wb = wt[buf];
#pragma unroll
    for (int nt = 0; nt < 4; ++nt) {
      v16h b = frag_ld(wb + (nt * 16 + col) * 32, h);
      acc0[nt] = wmma16(a0, b, acc0[nt]);
      acc1[nt] = wmma16(a1, b, acc1[nt]);
    }
    __syncthreads();
    buf ^= 1;
  }
#pragma unroll
  for (int sm = 0; sm < 2; ++sm) {
    const v8f* ac = sm ? acc1 : acc0;
#pragma unroll
    for (int nt = 0; nt < 4; ++nt)
#pragma unroll
      for (int v = 0; v < 8; ++v) {
        size_t idx =
            (size_t)(m0 + sm * 16 + v + 8 * h) * N + n0 + nt * 16 + col;
        float r = ac[nt][v];
        if (EPI == 0)      ((f16*)outp)[idx] = (f16)r;
        else if (EPI == 1) ((float*)outp)[idx] = r;
        else               ((float*)outp)[idx] = resid[idx] + r;
      }
  }
}

// ------------- fused gate_up GEMM + silu(gate)*up epilogue -------------------
// Stages 128 rows (64 gate + 64 up) per K-step; 16 WMMAs per wave per K-step.
__global__ void __launch_bounds__(256)
gateup_wmma_k(const f16* __restrict__ A, const f16* __restrict__ W,
              f16* __restrict__ act) {
  __shared__ __attribute__((aligned(16))) f16 wt[2][128 * 32];
  const int tid = threadIdx.x;
  const int lane = tid & 31, wave = tid >> 5;
  const int m0 = blockIdx.y * 256 + wave * 32;
  const int n0 = blockIdx.x * 64;
  const int col = lane & 15, h = lane >> 4;
  const f16* a_row0 = A + (size_t)(m0 + col) * HDIM;
  const f16* a_row1 = A + (size_t)(m0 + 16 + col) * HDIM;
  const int srow = tid >> 2, sseg = (tid & 3) * 8;
  const f16* g_src_g = W + (size_t)(n0 + srow) * HDIM + sseg;
  const f16* g_src_u = W + (size_t)(INTER + n0 + srow) * HDIM + sseg;

  v8f ag0[4] = {}, ag1[4] = {}, au0[4] = {}, au1[4] = {};
  cp16_g2l(&wt[0][srow * 32 + sseg], g_src_g);
  cp16_g2l(&wt[0][(64 + srow) * 32 + sseg], g_src_u);
  int buf = 0;
  for (int k0 = 0; k0 < HDIM; k0 += 32) {
    if (k0 + 32 < HDIM) {
      cp16_g2l(&wt[buf ^ 1][srow * 32 + sseg], g_src_g + k0 + 32);
      cp16_g2l(&wt[buf ^ 1][(64 + srow) * 32 + sseg], g_src_u + k0 + 32);
      cp_commit<2>();
    } else {
      cp_commit<0>();
    }
    __syncthreads();
    v16h a0 = frag_ld(a_row0 + k0, h);
    v16h a1 = frag_ld(a_row1 + k0, h);
    const f16* wb = wt[buf];
#pragma unroll
    for (int nt = 0; nt < 4; ++nt) {
      v16h bg = frag_ld(wb + (nt * 16 + col) * 32, h);
      v16h bu = frag_ld(wb + ((64 + nt * 16 + col)) * 32, h);
      ag0[nt] = wmma16(a0, bg, ag0[nt]);
      ag1[nt] = wmma16(a1, bg, ag1[nt]);
      au0[nt] = wmma16(a0, bu, au0[nt]);
      au1[nt] = wmma16(a1, bu, au1[nt]);
    }
    __syncthreads();
    buf ^= 1;
  }
#pragma unroll
  for (int sm = 0; sm < 2; ++sm) {
    const v8f* gg = sm ? ag1 : ag0;
    const v8f* uu = sm ? au1 : au0;
#pragma unroll
    for (int nt = 0; nt < 4; ++nt)
#pragma unroll
      for (int v = 0; v < 8; ++v) {
        size_t idx =
            (size_t)(m0 + sm * 16 + v + 8 * h) * INTER + n0 + nt * 16 + col;
        float g = gg[nt][v];
        float s = g / (1.f + __expf(-g));  // silu
        act[idx] = (f16)(s * uu[nt][v]);
      }
  }
}

// ---------------- RoPE on q_pe (interleaved pairs) ---------------------------
__global__ void __launch_bounds__(256)
rope_q_k(f16* __restrict__ q, const int* __restrict__ pos) {
  int idx = blockIdx.x * 256 + threadIdx.x;
  if (idx >= TTOK * NH * (DR / 2)) return;
  const int i = idx & 31;
  const int head = (idx >> 5) & 15;
  const int t = idx >> 9;
  f16* p = q + (size_t)t * (NH * DQK) + head * DQK + DN + 2 * i;
  float x1 = (float)p[0], x2 = (float)p[1];
  float invf = powf(10000.f, -(float)(2 * i) / (float)DR);
  float ang = (float)pos[t] * invf;
  float c = cosf(ang), s = sinf(ang);
  p[0] = (f16)(x1 * c - x2 * s);
  p[1] = (f16)(x2 * c + x1 * s);
}

// -------- assemble per-head K = [k_nope | rope(k_pe) bcast], V ---------------
__global__ void __launch_bounds__(256)
assemble_kv_k(const f16* __restrict__ kv, const float* __restrict__ latent,
              const int* __restrict__ pos, f16* __restrict__ K_,
              f16* __restrict__ V_) {
  __shared__ f16 kpe[DR];
  const int t = blockIdx.x;
  if (threadIdx.x < DR / 2) {
    const int i = threadIdx.x;
    float x1 = latent[(size_t)t * (KVLR + DR) + KVLR + 2 * i];
    float x2 = latent[(size_t)t * (KVLR + DR) + KVLR + 2 * i + 1];
    float invf = powf(10000.f, -(float)(2 * i) / (float)DR);
    float ang = (float)pos[t] * invf;
    float c = cosf(ang), s = sinf(ang);
    kpe[2 * i]     = (f16)(x1 * c - x2 * s);
    kpe[2 * i + 1] = (f16)(x2 * c + x1 * s);
  }
  __syncthreads();
  for (int idx = threadIdx.x; idx < NH * DN; idx += 256) {
    int hd = idx >> 7, d = idx & 127;
    K_[(size_t)t * (NH * DQK) + hd * DQK + d] =
        kv[(size_t)t * (NH * (DN + DV)) + hd * (DN + DV) + d];
  }
  for (int idx = threadIdx.x; idx < NH * DR; idx += 256) {
    int hd = idx >> 6, r = idx & 63;
    K_[(size_t)t * (NH * DQK) + hd * DQK + DN + r] = kpe[r];
  }
  for (int idx = threadIdx.x; idx < NH * DV; idx += 256) {
    int hd = idx >> 7, d = idx & 127;
    V_[(size_t)t * (NH * DV) + hd * DV + d] =
        kv[(size_t)t * (NH * (DN + DV)) + hd * (DN + DV) + DN + d];
  }
}

// ---------------- causal flash attention, 1 wave per (qtile, head) -----------
__global__ void __launch_bounds__(32)
flash_attn_k(const f16* __restrict__ Q, const f16* __restrict__ Km,
             const f16* __restrict__ Vm, f16* __restrict__ O) {
  __shared__ __attribute__((aligned(16))) f16 vt[DV * 32];  // V^T 128x32
  __shared__ __attribute__((aligned(16))) f16 pt[16 * 32];  // P 16x32
  const int qi = blockIdx.x, head = blockIdx.y;
  const int lane = threadIdx.x;
  const int col = lane & 15, h = lane >> 4;
  const int m0 = qi * 16;

  v16h qf[6];
  const f16* qrow = Q + (size_t)(m0 + col) * (NH * DQK) + head * DQK;
#pragma unroll
  for (int kk = 0; kk < 6; ++kk) qf[kk] = frag_ld(qrow + kk * 32, h);

  v8f acc[8] = {};
  float mr[8], lr[8];
#pragma unroll
  for (int v = 0; v < 8; ++v) { mr[v] = -3.0e38f; lr[v] = 0.f; }

  const int nblk = (m0 + 47) >> 5;  // ceil((m0+16)/32)
  for (int kb = 0; kb < nblk; ++kb) {
    const int key0 = kb << 5;
    // stage V^T (32 keys x 128 dims) into LDS, transposed
    for (int it = lane; it < 32 * 16; it += 32) {
      int key = it >> 4, d0 = (it & 15) * 8;
      v8h vv = *(const v8h*)(Vm + (size_t)(key0 + key) * (NH * DV) +
                             head * DV + d0);
#pragma unroll
      for (int j = 0; j < 8; ++j) vt[(d0 + j) * 32 + key] = vv[j];
    }
    // S = Q * K^T for two 16-key column tiles
    v8f s0 = {}, s1 = {};
    const f16* kr0 = Km + (size_t)(key0 + col) * (NH * DQK) + head * DQK;
    const f16* kr1 = Km + (size_t)(key0 + 16 + col) * (NH * DQK) + head * DQK;
#pragma unroll
    for (int kk = 0; kk < 6; ++kk) {
      s0 = wmma16(qf[kk], frag_ld(kr0 + kk * 32, h), s0);
      s1 = wmma16(qf[kk], frag_ld(kr1 + kk * 32, h), s1);
    }
    const bool diag = (key0 + 31 > m0);
#pragma unroll
    for (int v = 0; v < 8; ++v) {
      const int r = m0 + v + 8 * h;
      float x0 = s0[v] * SCALEF, x1 = s1[v] * SCALEF;
      if (diag) {
        if (key0 + col > r)      x0 = -3.0e38f;
        if (key0 + 16 + col > r) x1 = -3.0e38f;
      }
      float t = fmaxf(x0, x1);
      t = fmaxf(t, __shfl_xor(t, 1, 16));
      t = fmaxf(t, __shfl_xor(t, 2, 16));
      t = fmaxf(t, __shfl_xor(t, 4, 16));
      t = fmaxf(t, __shfl_xor(t, 8, 16));
      float mnew = fmaxf(mr[v], t);
      float scl = __expf(mr[v] - mnew);
      float p0 = __expf(x0 - mnew), p1 = __expf(x1 - mnew);
      float ps = p0 + p1;
      ps += __shfl_xor(ps, 1, 16);
      ps += __shfl_xor(ps, 2, 16);
      ps += __shfl_xor(ps, 4, 16);
      ps += __shfl_xor(ps, 8, 16);
      lr[v] = lr[v] * scl + ps;
      mr[v] = mnew;
#pragma unroll
      for (int nt = 0; nt < 8; ++nt) acc[nt][v] *= scl;
      pt[(v + 8 * h) * 32 + col]      = (f16)p0;  // D-layout -> LDS
      pt[(v + 8 * h) * 32 + 16 + col] = (f16)p1;
    }
    asm volatile("s_wait_dscnt 0x0" ::: "memory");
    // O += P * V  (A from LDS in A-layout, B from transposed V in LDS)
    v16h pf = frag_ld(pt + (size_t)col * 32, h);
#pragma unroll
    for (int nt = 0; nt < 8; ++nt) {
      v16h bf = frag_ld(vt + (size_t)(nt * 16 + col) * 32, h);
      acc[nt] = wmma16(pf, bf, acc[nt]);
    }
    asm volatile("s_wait_dscnt 0x0" ::: "memory");
  }
#pragma unroll
  for (int nt = 0; nt < 8; ++nt)
#pragma unroll
    for (int v = 0; v < 8; ++v)
      O[(size_t)(m0 + v + 8 * h) * (NH * DV) + head * DV + nt * 16 + col] =
          (f16)(acc[nt][v] / lr[v]);
}

// ---------------- host-side orchestration ------------------------------------
extern "C" void kernel_launch(void* const* d_in, const int* in_sizes, int n_in,
                              void* d_out, int out_size, void* d_ws,
                              size_t ws_size, hipStream_t stream) {
  (void)in_sizes; (void)n_in; (void)out_size; (void)ws_size;
  const float* hidden     = (const float*)d_in[0];
  const int*   positions  = (const int*)d_in[1];
  const float* input_ln_w = (const float*)d_in[2];
  const float* post_ln_w  = (const float*)d_in[3];
  const float* q_a_w      = (const float*)d_in[4];
  const float* q_a_ln_w   = (const float*)d_in[5];
  const float* q_b_w      = (const float*)d_in[6];
  const float* kv_a_w     = (const float*)d_in[7];
  const float* kv_a_ln_w  = (const float*)d_in[8];
  const float* kv_b_w     = (const float*)d_in[9];
  const float* o_w        = (const float*)d_in[10];
  const float* gate_up_w  = (const float*)d_in[11];
  const float* down_w     = (const float*)d_in[12];

  char* p = (char*)d_ws;
  auto alloc_h = [&](size_t n) -> f16* {
    f16* r = (f16*)p; p += ((n * sizeof(f16)) + 255) & ~(size_t)255; return r;
  };
  auto alloc_f = [&](size_t n) -> float* {
    float* r = (float*)p; p += ((n * sizeof(float)) + 255) & ~(size_t)255; return r;
  };

  // f16 weights
  f16* w_qa  = alloc_h((size_t)QLR * HDIM);
  f16* w_qb  = alloc_h((size_t)NH * DQK * QLR);
  f16* w_kva = alloc_h((size_t)(KVLR + DR) * HDIM);
  f16* w_kvb = alloc_h((size_t)NH * (DN + DV) * KVLR);
  f16* w_o   = alloc_h((size_t)HDIM * NH * DV);
  f16* w_gu  = alloc_h((size_t)2 * INTER * HDIM);
  f16* w_dn  = alloc_h((size_t)HDIM * INTER);
  // f16 activations
  f16* hsn   = alloc_h((size_t)TTOK * HDIM);
  f16* qa    = alloc_h((size_t)TTOK * QLR);
  f16* qc    = alloc_h((size_t)TTOK * QLR);
  f16* qh    = alloc_h((size_t)TTOK * NH * DQK);
  f16* kva   = alloc_h((size_t)TTOK * KVLR);
  f16* kvh   = alloc_h((size_t)TTOK * NH * (DN + DV));
  f16* kh    = alloc_h((size_t)TTOK * NH * DQK);
  f16* vh    = alloc_h((size_t)TTOK * NH * DV);
  f16* attnh = alloc_h((size_t)TTOK * NH * DV);
  f16* hs2n  = alloc_h((size_t)TTOK * HDIM);
  f16* acth  = alloc_h((size_t)TTOK * INTER);
  // f32
  float* latent = alloc_f((size_t)TTOK * (KVLR + DR));
  float* resid2 = alloc_f((size_t)TTOK * HDIM);

  auto cvt = [&](const float* s, f16* d, size_t n) {
    cvt_f32_f16<<<2048, 256, 0, stream>>>(s, d, (long long)n);
  };
  cvt(q_a_w, w_qa, (size_t)QLR * HDIM);
  cvt(q_b_w, w_qb, (size_t)NH * DQK * QLR);
  cvt(kv_a_w, w_kva, (size_t)(KVLR + DR) * HDIM);
  cvt(kv_b_w, w_kvb, (size_t)NH * (DN + DV) * KVLR);
  cvt(o_w, w_o, (size_t)HDIM * NH * DV);
  cvt(gate_up_w, w_gu, (size_t)2 * INTER * HDIM);
  cvt(down_w, w_dn, (size_t)HDIM * INTER);

  // 1) input RMSNorm
  rmsnorm_k<float><<<TTOK, 256, 0, stream>>>(hidden, input_ln_w, hsn, HDIM, HDIM);
  // 2) q_a = hsn @ q_a_w^T ; q_a RMSNorm
  gemm_wmma_k<0><<<dim3(QLR / 64, TTOK / 256), 256, 0, stream>>>(
      hsn, w_qa, (void*)qa, nullptr, QLR, HDIM);
  rmsnorm_k<f16><<<TTOK, 256, 0, stream>>>(qa, q_a_ln_w, qc, QLR, QLR);
  // 3) q = qc @ q_b_w^T ; RoPE on q_pe
  gemm_wmma_k<0><<<dim3(NH * DQK / 64, TTOK / 256), 256, 0, stream>>>(
      qc, w_qb, (void*)qh, nullptr, NH * DQK, QLR);
  rope_q_k<<<(TTOK * NH * (DR / 2) + 255) / 256, 256, 0, stream>>>(qh, positions);
  // 4) latent = hsn @ kv_a_w^T (f32, keeps raw k_pe); kv_a RMSNorm
  gemm_wmma_k<1><<<dim3((KVLR + DR) / 64, TTOK / 256), 256, 0, stream>>>(
      hsn, w_kva, (void*)latent, nullptr, KVLR + DR, HDIM);
  rmsnorm_k<float><<<TTOK, 256, 0, stream>>>(latent, kv_a_ln_w, kva, KVLR,
                                             KVLR + DR);
  // 5) kv = kva @ kv_b_w^T ; assemble per-head K (with RoPE'd k_pe) and V
  gemm_wmma_k<0><<<dim3(NH * (DN + DV) / 64, TTOK / 256), 256, 0, stream>>>(
      kva, w_kvb, (void*)kvh, nullptr, NH * (DN + DV), KVLR);
  assemble_kv_k<<<TTOK, 256, 0, stream>>>(kvh, latent, positions, kh, vh);
  // 6) causal flash attention
  flash_attn_k<<<dim3(TTOK / 16, NH), 32, 0, stream>>>(qh, kh, vh, attnh);
  // 7) o-proj + residual
  gemm_wmma_k<2><<<dim3(HDIM / 64, TTOK / 256), 256, 0, stream>>>(
      attnh, w_o, (void*)resid2, hidden, HDIM, NH * DV);
  // 8) post RMSNorm
  rmsnorm_k<float><<<TTOK, 256, 0, stream>>>(resid2, post_ln_w, hs2n, HDIM, HDIM);
  // 9) fused gate_up + silu*up
  gateup_wmma_k<<<dim3(INTER / 64, TTOK / 256), 256, 0, stream>>>(hs2n, w_gu,
                                                                  acth);
  // 10) down-proj + residual -> output (f32)
  gemm_wmma_k<2><<<dim3(HDIM / 64, TTOK / 256), 256, 0, stream>>>(
      acth, w_dn, d_out, resid2, HDIM, INTER);
}
